// GRUDModel_38903813767398
// MI455X (gfx1250) — matrix operand order repaired and checked by reading, not compiled, not run on hardware
//
#include <hip/hip_runtime.h>
#include <hip/hip_bf16.h>

typedef __attribute__((ext_vector_type(16))) _Float16 v16h;
typedef __attribute__((ext_vector_type(8)))  _Float16 v8h;
typedef __attribute__((ext_vector_type(8)))  float    v8f;
typedef __attribute__((ext_vector_type(4)))  unsigned int v4u;
typedef __attribute__((ext_vector_type(8)))  int      v8i_;
typedef __attribute__((ext_vector_type(4)))  int      v4i_;

#define B_ 256
#define T_ 200
#define D_ 128
#define H_ 256

// ---------------------------------------------------------------------------
// WMMA helpers (gfx1250: V_WMMA_F32_16X16X32_F16, wave32)
// ---------------------------------------------------------------------------
__device__ __forceinline__ v8f wmma_f16(v16h a, v16h b, v8f c) {
  return __builtin_amdgcn_wmma_f32_16x16x32_f16(false, a, false, b,
                                                (short)0, c, false, false);
}

// A-operand loader: per-lane 16 f16 = {K..K+7} in elems 0-7, {K+8..K+15} in 8-15.
__device__ __forceinline__ v16h ldA16(const _Float16* p) {
  v8h lo = *(const v8h*)(p);
  v8h hi = *(const v8h*)(p + 8);
  v16h a;
#pragma unroll
  for (int i = 0; i < 8; ++i) { a[i] = lo[i]; a[i + 8] = hi[i]; }
  return a;
}

__device__ __forceinline__ float hardsig(float x) {
  return fminf(fmaxf(x * (1.0f / 6.0f) + 0.5f, 0.0f), 1.0f);
}

// ---------------------------------------------------------------------------
// TDM: 1-D DMA of `bytes` (multiple of 8) from global -> LDS. D# per ISA ch.8.
// data_size=8B; tensor_dim0 = tile_dim0 = stride = bytes/8; type=2 ("image").
// Tracked on TENSORcnt; pair with s_wait_tensorcnt.
// This toolchain exposes the 6-arg builtin: (g0, g1, g2, g3, g_extra8, cpol).
// ---------------------------------------------------------------------------
__device__ __forceinline__ void tdm_copy_1d(unsigned lds_byte_addr,
                                            const void* gptr, unsigned bytes) {
  unsigned long long ga = (unsigned long long)(size_t)gptr;
  unsigned units = bytes >> 3;  // 8-byte elements
  v4u g0;
  g0[0] = 1u;                                    // count=1 (valid user D#)
  g0[1] = lds_byte_addr;                         // lds_addr (bytes)
  g0[2] = (unsigned)ga;                          // global_addr[31:0]
  g0[3] = (unsigned)(ga >> 32) | (2u << 30);     // global_addr[56:32] | type=2
  v8i_ g1;
  g1[0] = (3 << 16);                             // data_size = 8B
  g1[1] = (int)((units & 0xFFFFu) << 16);        // tensor_dim0[15:0]
  g1[2] = (int)(((units >> 16) & 0xFFFFu) | (1u << 16)); // dim0[31:16] | tensor_dim1=1
  g1[3] = (int)((units & 0xFFFFu) << 16);        // tile_dim0
  g1[4] = 0;                                     // tile_dim1 = tile_dim2 = 0 (unused)
  g1[5] = (int)units;                            // tensor_dim0_stride[31:0]
  g1[6] = 0;
  g1[7] = 0;
  v4i_ gz4 = {0, 0, 0, 0};
  v8i_ gz8 = {0, 0, 0, 0, 0, 0, 0, 0};
  __builtin_amdgcn_tensor_load_to_lds(g0, g1, gz4, gz4, gz8, 0);
}

// ---------------------------------------------------------------------------
// Kernel 0a: fp32 -> fp16 weight conversion (layout preserved, already (N,K))
// ---------------------------------------------------------------------------
__global__ void conv_f16(const float* __restrict__ src, _Float16* __restrict__ dst, int n) {
  int i = blockIdx.x * blockDim.x + threadIdx.x;
  if (i < n) dst[i] = (_Float16)src[i];
}

// Kernel 0b: transpose (R x C) fp32 -> (C x R) fp16  (for hidden_decay_kernel)
__global__ void transpose_f16(const float* __restrict__ src, _Float16* __restrict__ dst,
                              int R, int C) {
  int i = blockIdx.x * blockDim.x + threadIdx.x;
  if (i < R * C) {
    int r = i / C, c = i % C;
    dst[(size_t)c * R + r] = (_Float16)src[i];
  }
}

// ---------------------------------------------------------------------------
// Kernel 1: per-(b,d) elementwise scan over T (mask carry + time gaps).
// ---------------------------------------------------------------------------
__global__ void grud_scan(const float* __restrict__ x, const unsigned char* __restrict__ m,
                          const float* __restrict__ s,
                          const float* __restrict__ idk, const float* __restrict__ idb,
                          _Float16* __restrict__ xt, _Float16* __restrict__ m1,
                          _Float16* __restrict__ dg) {
  int gid = blockIdx.x * blockDim.x + threadIdx.x;
  if (gid >= B_ * D_) return;
  int b = gid / D_, d = gid % D_;
  const float k = idk[d], kb = idb[d];
  float s_prev = 0.f, x_keep = 0.f;
  for (int t = 0; t < T_; ++t) {
    size_t ii = ((size_t)b * T_ + t) * D_ + d;     // inputs are [B,T,D]
    float xv = x[ii];
    bool  mv = m[ii] != 0;
    float sv = s[ii];
    float dd = sv - s_prev;
    float gdi = __expf(-fmaxf(dd * k + kb, 0.f));  // exp(-relu(.))
    float xk  = mv ? xv : x_keep;
    float xtv = mv ? xv : gdi * xk;
    x_keep = xk;
    s_prev = mv ? sv : s_prev;
    size_t wi = ((size_t)t * B_ + b) * D_ + d;     // ws is [T,B,D]
    xt[wi] = (_Float16)xtv;
    m1[wi] = (_Float16)(mv ? 0.f : 1.f);
    dg[wi] = (_Float16)dd;
  }
}

// ---------------------------------------------------------------------------
// Kernel 2: hoisted input projections for ALL (t,b) rows (parallel WMMA).
// ---------------------------------------------------------------------------
__global__ __launch_bounds__(256) void grud_gemm_in(
    const _Float16* __restrict__ xt, const _Float16* __restrict__ m1,
    const _Float16* __restrict__ dg,
    const _Float16* __restrict__ wz, const _Float16* __restrict__ wr,
    const _Float16* __restrict__ wh,
    const _Float16* __restrict__ mz, const _Float16* __restrict__ mr,
    const _Float16* __restrict__ mh,
    const _Float16* __restrict__ wdhT,
    const float* __restrict__ bz, const float* __restrict__ br,
    const float* __restrict__ bh, const float* __restrict__ bdh,
    float* __restrict__ prez, float* __restrict__ prer,
    float* __restrict__ preh, float* __restrict__ gdh) {
  const int Mtiles = (T_ * B_) / 16;  // 3200
  const int Ntiles = H_ / 16;         // 16
  int wid = blockIdx.x * (blockDim.x >> 5) + (threadIdx.x >> 5);
  if (wid >= Mtiles * Ntiles * 4) return;
  int g   = wid / (Mtiles * Ntiles);
  int rem = wid % (Mtiles * Ntiles);
  int nt  = rem % Ntiles;
  int mt  = rem / Ntiles;

  int lane  = threadIdx.x & 31;
  int arow  = lane & 15, akoff = (lane >> 4) * 16;
  int bcol  = lane & 15, bkoff = (lane >> 4) * 16;
  int crow0 = (lane >> 4) * 8, cn = lane & 15;

  const _Float16 *A0, *B0, *A1 = nullptr, *B1 = nullptr;
  const float* bias;
  float* outp;
  switch (g) {
    case 0:  A0 = xt; B0 = wz;   A1 = m1; B1 = mz; bias = bz;  outp = prez; break;
    case 1:  A0 = xt; B0 = wr;   A1 = m1; B1 = mr; bias = br;  outp = prer; break;
    case 2:  A0 = xt; B0 = wh;   A1 = m1; B1 = mh; bias = bh;  outp = preh; break;
    default: A0 = dg; B0 = wdhT;                   bias = bdh; outp = gdh;  break;
  }

  size_t r0 = (size_t)mt * 16;
  int    n0 = nt * 16;
  v8f acc = {0.f, 0.f, 0.f, 0.f, 0.f, 0.f, 0.f, 0.f};

#pragma unroll
  for (int k0 = 0; k0 < D_; k0 += 32) {
    v16h a = ldA16(A0 + (r0 + arow) * D_ + k0 + akoff);
    v16h b = *(const v16h*)(B0 + (size_t)(n0 + bcol) * D_ + k0 + bkoff);
    acc = wmma_f16(a, b, acc);
  }
  if (g < 3) {
#pragma unroll
    for (int k0 = 0; k0 < D_; k0 += 32) {
      v16h a = ldA16(A1 + (r0 + arow) * D_ + k0 + akoff);
      v16h b = *(const v16h*)(B1 + (size_t)(n0 + bcol) * D_ + k0 + bkoff);
      acc = wmma_f16(a, b, acc);
    }
  }

  float bv = bias[n0 + cn];
#pragma unroll
  for (int j = 0; j < 8; ++j) {
    float v = acc[j] + bv;
    if (g == 3) v = __expf(-fmaxf(v, 0.f));
    outp[(r0 + crow0 + j) * H_ + n0 + cn] = v;
  }
}

// ---------------------------------------------------------------------------
// Kernel 3: recurrence. 16 persistent WGs (one per 16-row batch tile).
// h in LDS fp32; A-operands fp16 in LDS; per-step activations (gdh/pre_z/r/h)
// TDM double-buffered into LDS; U matrices streamed from L2 as B-operands.
// Dynamic LDS layout:
//   [0      .. 16384)  h        16x256 f32
//   [16384  .. 24832)  hd16     16xLDH f16
//   [24832  .. 33280)  rh16     16xLDH f16
//   [33280  .. 164352) act      2 bufs x 4 arrays x (16x256 f32 = 16 KB)
// ---------------------------------------------------------------------------
#define LDH        (H_ + 8)
#define TILE_F32   (16 * H_)          // 4096 floats = 16 KB
#define OFF_HD     16384
#define OFF_RH     24832
#define OFF_ACT    33280u
#define SMEM_REC   (OFF_ACT + 2 * 4 * TILE_F32 * 4)   // 164352 bytes

__global__ __launch_bounds__(256, 1) void grud_rec(
    const _Float16* __restrict__ uz, const _Float16* __restrict__ ur,
    const _Float16* __restrict__ uh,
    const float* __restrict__ prez, const float* __restrict__ prer,
    const float* __restrict__ preh, const float* __restrict__ gdh,
    float* __restrict__ out) {
  extern __shared__ char smem[];
  float*    h_lds = (float*)smem;
  _Float16* hd16  = (_Float16*)(smem + OFF_HD);
  _Float16* rh16  = (_Float16*)(smem + OFF_RH);
  float*    act   = (float*)(smem + OFF_ACT);
  const unsigned actLdsBase = (unsigned)(size_t)act;  // generic addr low 32 = LDS offset

  const int tid  = threadIdx.x;
  const int b0   = blockIdx.x * 16;
  const int wave = tid >> 5, lane = tid & 31;

  const int arow  = lane & 15, akoff = (lane >> 4) * 16;
  const int bcol  = lane & 15, bkoff = (lane >> 4) * 16;
  const int crow0 = (lane >> 4) * 8, cn = lane & 15;

  // Prologue: waves 0-3 kick off TDM for t=0 tiles (buffer 0).
  if (wave < 4) {
    const float* src = (wave == 0) ? gdh : (wave == 1) ? prez
                     : (wave == 2) ? prer : preh;
    tdm_copy_1d(actLdsBase + (unsigned)wave * (TILE_F32 * 4),
                src + (size_t)b0 * H_, TILE_F32 * 4);
  }
  for (int i = tid; i < 16 * H_; i += 256) h_lds[i] = 0.f;

  for (int t = 0; t < T_; ++t) {
    __builtin_amdgcn_s_wait_tensorcnt(0);   // buffer (t&1) filled (no-op for non-issuers)
    __syncthreads();

    const size_t rowbase = (size_t)t * B_ + b0;
    const int cur = t & 1;

    // Kick off TDM for step t+1 into the other buffer (last read at step t-1).
    if (wave < 4 && t + 1 < T_) {
      const float* src = (wave == 0) ? gdh : (wave == 1) ? prez
                       : (wave == 2) ? prer : preh;
      tdm_copy_1d(actLdsBase + (unsigned)(((t + 1) & 1) * 4 + wave) * (TILE_F32 * 4),
                  src + (rowbase + B_) * H_, TILE_F32 * 4);
    }

    const float* agdh = act + (cur * 4 + 0) * TILE_F32;
    const float* apz  = act + (cur * 4 + 1) * TILE_F32;
    const float* apr  = act + (cur * 4 + 2) * TILE_F32;
    const float* aph  = act + (cur * 4 + 3) * TILE_F32;

    // Phase A: h_d = gamma_dh * h  -> fp16 LDS (WMMA A operand)
    for (int i = tid; i < 16 * H_; i += 256) {
      int r = i >> 8, c = i & (H_ - 1);
      hd16[r * LDH + c] = (_Float16)(agdh[i] * h_lds[i]);
    }
    __syncthreads();

    // Phase B: z, r gates for this wave's 32 columns
    float zv[2][8];
#pragma unroll
    for (int ntile = 0; ntile < 2; ++ntile) {
      int n0 = wave * 32 + ntile * 16;
      v8f cz, cr;
#pragma unroll
      for (int j = 0; j < 8; ++j) {
        int o = (crow0 + j) * H_ + n0 + cn;
        cz[j] = apz[o];
        cr[j] = apr[o];
      }
#pragma unroll
      for (int k0 = 0; k0 < H_; k0 += 32) {
        v16h a  = ldA16(hd16 + arow * LDH + k0 + akoff);
        v16h bz = *(const v16h*)(uz + (size_t)(n0 + bcol) * H_ + k0 + bkoff);
        v16h br = *(const v16h*)(ur + (size_t)(n0 + bcol) * H_ + k0 + bkoff);
        cz = wmma_f16(a, bz, cz);
        cr = wmma_f16(a, br, cr);
      }
#pragma unroll
      for (int j = 0; j < 8; ++j) {
        float z = hardsig(cz[j]);
        float r = hardsig(cr[j]);
        zv[ntile][j] = z;
        int mrow = crow0 + j;
        rh16[mrow * LDH + n0 + cn] =
            (_Float16)(r * (float)hd16[mrow * LDH + n0 + cn]);
      }
    }
    __syncthreads();

    // Phase C: candidate hh, blend, write back h + output
#pragma unroll
    for (int ntile = 0; ntile < 2; ++ntile) {
      int n0 = wave * 32 + ntile * 16;
      v8f ch;
#pragma unroll
      for (int j = 0; j < 8; ++j)
        ch[j] = aph[(crow0 + j) * H_ + n0 + cn];
#pragma unroll
      for (int k0 = 0; k0 < H_; k0 += 32) {
        v16h a  = ldA16(rh16 + arow * LDH + k0 + akoff);
        v16h bh = *(const v16h*)(uh + (size_t)(n0 + bcol) * H_ + k0 + bkoff);
        ch = wmma_f16(a, bh, ch);
      }
#pragma unroll
      for (int j = 0; j < 8; ++j) {
        float hh = tanhf(ch[j]);
        int   mr = crow0 + j, nn = n0 + cn;
        float z  = zv[ntile][j];
        float hn = z * h_lds[mr * H_ + nn] + (1.f - z) * hh;
        h_lds[mr * H_ + nn] = hn;
        out[((size_t)(b0 + mr) * T_ + t) * H_ + nn] = hn;   // [B,T,H]
      }
    }
    __syncthreads();
  }
}

// ---------------------------------------------------------------------------
// Host launcher
// ---------------------------------------------------------------------------
extern "C" void kernel_launch(void* const* d_in, const int* in_sizes, int n_in,
                              void* d_out, int out_size, void* d_ws, size_t ws_size,
                              hipStream_t stream) {
  const float*         x   = (const float*)d_in[0];
  const unsigned char* m   = (const unsigned char*)d_in[1];   // jnp bool = 1 byte
  const float*         s   = (const float*)d_in[2];
  const float* Wz  = (const float*)d_in[3];
  const float* Wr  = (const float*)d_in[4];
  const float* Wh  = (const float*)d_in[5];
  const float* Uz  = (const float*)d_in[6];
  const float* Ur  = (const float*)d_in[7];
  const float* Uh  = (const float*)d_in[8];
  const float* bz  = (const float*)d_in[9];
  const float* br  = (const float*)d_in[10];
  const float* bh  = (const float*)d_in[11];
  const float* idk = (const float*)d_in[12];
  const float* idb = (const float*)d_in[13];
  const float* Wdh = (const float*)d_in[14];  // (D,H)
  const float* bdh = (const float*)d_in[15];
  const float* Mz  = (const float*)d_in[16];
  const float* Mr  = (const float*)d_in[17];
  const float* Mh  = (const float*)d_in[18];

  char* ws = (char*)d_ws;
  auto carve = [&](size_t bytes) -> char* {
    char* p = ws;
    ws += (bytes + 255) & ~(size_t)255;
    return p;
  };
  const size_t HD = (size_t)H_ * D_, HH = (size_t)H_ * H_;
  const size_t ROWS = (size_t)T_ * B_;

  _Float16* wz16   = (_Float16*)carve(HD * 2);
  _Float16* wr16   = (_Float16*)carve(HD * 2);
  _Float16* wh16   = (_Float16*)carve(HD * 2);
  _Float16* mz16   = (_Float16*)carve(HD * 2);
  _Float16* mr16   = (_Float16*)carve(HD * 2);
  _Float16* mh16   = (_Float16*)carve(HD * 2);
  _Float16* uz16   = (_Float16*)carve(HH * 2);
  _Float16* ur16   = (_Float16*)carve(HH * 2);
  _Float16* uh16   = (_Float16*)carve(HH * 2);
  _Float16* wdhT16 = (_Float16*)carve(HD * 2);
  _Float16* xt16   = (_Float16*)carve(ROWS * D_ * 2);
  _Float16* m116   = (_Float16*)carve(ROWS * D_ * 2);
  _Float16* dg16   = (_Float16*)carve(ROWS * D_ * 2);
  float*    prez   = (float*)carve(ROWS * H_ * 4);
  float*    prer   = (float*)carve(ROWS * H_ * 4);
  float*    preh   = (float*)carve(ROWS * H_ * 4);
  float*    gdh    = (float*)carve(ROWS * H_ * 4);

  // Phase 0: weight conversion / transpose
  int nHD = (int)HD, nHH = (int)HH;
  conv_f16<<<(nHD + 255) / 256, 256, 0, stream>>>(Wz, wz16, nHD);
  conv_f16<<<(nHD + 255) / 256, 256, 0, stream>>>(Wr, wr16, nHD);
  conv_f16<<<(nHD + 255) / 256, 256, 0, stream>>>(Wh, wh16, nHD);
  conv_f16<<<(nHD + 255) / 256, 256, 0, stream>>>(Mz, mz16, nHD);
  conv_f16<<<(nHD + 255) / 256, 256, 0, stream>>>(Mr, mr16, nHD);
  conv_f16<<<(nHD + 255) / 256, 256, 0, stream>>>(Mh, mh16, nHD);
  conv_f16<<<(nHH + 255) / 256, 256, 0, stream>>>(Uz, uz16, nHH);
  conv_f16<<<(nHH + 255) / 256, 256, 0, stream>>>(Ur, ur16, nHH);
  conv_f16<<<(nHH + 255) / 256, 256, 0, stream>>>(Uh, uh16, nHH);
  transpose_f16<<<(nHD + 255) / 256, 256, 0, stream>>>(Wdh, wdhT16, D_, H_);

  // Phase 1: mask/time scan
  grud_scan<<<(B_ * D_ + 255) / 256, 256, 0, stream>>>(x, m, s, idk, idb,
                                                       xt16, m116, dg16);

  // Phase 2: hoisted input projections + decay GEMMs (WMMA)
  {
    int Mtiles = (int)(ROWS / 16), Ntiles = H_ / 16;
    int waves = Mtiles * Ntiles * 4;
    int blocks = (waves + 7) / 8;
    grud_gemm_in<<<blocks, 256, 0, stream>>>(xt16, m116, dg16,
                                             wz16, wr16, wh16,
                                             mz16, mr16, mh16, wdhT16,
                                             bz, br, bh, bdh,
                                             prez, prer, preh, gdh);
  }

  // Phase 3: recurrence (WMMA + TDM double-buffering), 16 batch tiles
  (void)hipFuncSetAttribute((const void*)grud_rec,
                            hipFuncAttributeMaxDynamicSharedMemorySize, SMEM_REC);
  grud_rec<<<B_ / 16, 256, SMEM_REC, stream>>>(uz16, ur16, uh16,
                                               prez, prer, preh, gdh,
                                               (float*)d_out);
  (void)in_sizes; (void)n_in; (void)out_size; (void)ws_size;
}
